// MultiverseEnsemble_69200513073191
// MI455X (gfx1250) — matrix operand order repaired and checked by reading, not compiled, not run on hardware
//
#include <hip/hip_runtime.h>
#include <hip/hip_bf16.h>

// MI455X / gfx1250, wave32. bf16 WMMA (v_wmma_f32_16x16x32_bf16) for all large GEMMs.

typedef __attribute__((ext_vector_type(16))) __bf16 v16bf;
typedef __attribute__((ext_vector_type(8)))  float  v8f;

struct alignas(16) Q16 { unsigned int w[4]; };
union FragBF { v16bf v; Q16 q[2]; };

#define U_ 128
#define B_ 1024
#define E_ 64

// ---------------- pack kernels: f32 -> bf16 in WMMA fragment layouts ----------------
// A-matrix 16x32 bf16 layout (ISA 7.12.2): lane = (hi<<4)|m ; elems 0..7 -> K=hi*8+e,
// elems 8..15 -> K=16+hi*8+(e-8). Stored 16 contiguous bf16 per lane, 512 per fragment.
__global__ __launch_bounds__(256)
void pack_x_kernel(const float* __restrict__ x, __bf16* __restrict__ out)
{
    const int idx = blockIdx.x * 256 + threadIdx.x;       // 64 rt * 8 kt * 512
    if (idx >= 64 * 8 * 512) return;
    const int frag = idx >> 9, r2 = idx & 511;
    const int lane = r2 >> 4, e = r2 & 15;
    const int rt = frag >> 3, kt = frag & 7;
    const int lm = lane & 15, lh = lane >> 4;
    const int row = rt * 16 + lm;
    const int k = kt * 32 + ((e < 8) ? (lh * 8 + e) : (16 + lh * 8 + (e - 8)));
    out[idx] = (__bf16)x[row * 256 + k];
}

// B-matrix 32x16 bf16 layout: lane = (hi<<4)|n ; elems 0..7 -> K = hi*8 + e, elems 8..15 -> K = 16 + hi*8 + (e-8).
__global__ __launch_bounds__(256)
void pack_w_kernel(const float* __restrict__ w, __bf16* __restrict__ out,
                   const int K, const int N)
{
    const int u   = blockIdx.y;
    const int per = (K >> 5) * (N >> 4) * 512;
    const int idx = blockIdx.x * 256 + threadIdx.x;
    if (idx >= per) return;
    const int frag = idx >> 9, r2 = idx & 511;
    const int lane = r2 >> 4, e = r2 & 15;
    const int nnt = N >> 4;
    const int kt = frag / nnt, nt = frag % nnt;
    const int kk = kt * 32 + ((e < 8) ? ((lane >> 4) * 8 + e) : (16 + (lane >> 4) * 8 + (e - 8)));
    const int nn = nt * 16 + (lane & 15);
    out[(size_t)u * per + idx] = (__bf16)w[(size_t)u * K * N + (size_t)kk * N + nn];
}

// ---------------- fused per-universe MLP: 3 WMMA GEMM stages + LN + small head --------
__global__ __launch_bounds__(256)
void universe_kernel(
    const float* __restrict__ fs,  const float* __restrict__ dm,
    const float* __restrict__ b1,  const float* __restrict__ g1,  const float* __restrict__ be1,
    const float* __restrict__ b2,  const float* __restrict__ g2,  const float* __restrict__ be2,
    const float* __restrict__ b3,
    const float* __restrict__ Wr,  const float* __restrict__ br,
    const float* __restrict__ Wa1, const float* __restrict__ ba1,
    const float* __restrict__ Wa2, const float* __restrict__ ba2,
    const float* __restrict__ Wa3, const float* __restrict__ ba3,
    const __bf16* __restrict__ xpack, const __bf16* __restrict__ w1p,
    const __bf16* __restrict__ w2p,   const __bf16* __restrict__ w3p,
    float* __restrict__ encG, float* __restrict__ scoresG)
{
    extern __shared__ char smem[];
    float*  fbuf  = (float*)smem;                  // [32][256] f32 (reused as [32][128], [32][64])
    __bf16* bbuf  = (__bf16*)(smem + 32768);       // [32][256] bf16 (reused as [32][128])
    float*  tbuf  = (float*)(smem + 49152);        // [32][64]
    float*  a1buf = tbuf + 32 * 64;                // [32][32]
    float*  a2buf = a1buf + 32 * 32;               // [32][16]
    float*  ssbuf = a2buf + 32 * 16;               // [32]

    const int u    = blockIdx.x;
    const int bb   = blockIdx.y;
    const int b0   = bb * 32;
    const int tid  = threadIdx.x;
    const int wave = tid >> 5;
    const int lane = tid & 31;
    const int lm   = lane & 15;
    const int lh   = lane >> 4;
    const int rt   = wave & 1;      // row-tile (2 x 16 rows)
    const int cq   = wave >> 1;     // column quarter
    const float corr = fs[u] * 137.036f;

    // ---- stage 1: h1 = relu(LN((x @ W1)*corr + b1))  [32 x 256]
    {
        const int xrt = bb * 2 + rt;
        for (int nt0 = 0; nt0 < 4; ++nt0) {
            const int nt = cq * 4 + nt0;
            v8f c = {};
            #pragma unroll
            for (int kt = 0; kt < 8; ++kt) {
                FragBF a, b;
                a.v = *(const v16bf*)(xpack + (size_t)(xrt * 8 + kt) * 512 + lane * 16);
                b.v = *(const v16bf*)(w1p + (size_t)((u * 8 + kt) * 16 + nt) * 512 + lane * 16);
                __builtin_prefetch(w1p + (size_t)((u * 8 + kt + 1) * 16 + nt) * 512 + lane * 16, 0, 1);
                c = __builtin_amdgcn_wmma_f32_16x16x32_bf16(false, a.v, false, b.v,
                                                            (short)0, c, false, false);
            }
            const int col = nt * 16 + lm;
            #pragma unroll
            for (int j = 0; j < 8; ++j)
                fbuf[(rt * 16 + lh * 8 + j) * 256 + col] = c[j] * corr + b1[u * 256 + col];
        }
    }
    __syncthreads();
    for (int rr = 0; rr < 4; ++rr) {                 // LN over 256 per row (ddof=0)
        const int row = wave * 4 + rr;
        float vv[8];
        #pragma unroll
        for (int e = 0; e < 8; ++e) vv[e] = fbuf[row * 256 + lane * 8 + e];
        float s = 0.f, s2 = 0.f;
        #pragma unroll
        for (int e = 0; e < 8; ++e) { s += vv[e]; s2 += vv[e] * vv[e]; }
        #pragma unroll
        for (int off = 16; off > 0; off >>= 1) {
            s += __shfl_xor(s, off, 32); s2 += __shfl_xor(s2, off, 32);
        }
        const float mean = s * (1.f / 256.f);
        const float rstd = rsqrtf(s2 * (1.f / 256.f) - mean * mean + 1e-5f);
        #pragma unroll
        for (int e = 0; e < 8; ++e) {
            const int col = lane * 8 + e;
            const float t = (vv[e] - mean) * rstd * g1[u * 256 + col] + be1[u * 256 + col];
            bbuf[row * 256 + col] = (__bf16)fmaxf(t, 0.f);
        }
    }
    __syncthreads();

    // ---- stage 2: h2 = relu(LN(h1 @ W2 + b2))  [32 x 128]
    for (int nt0 = 0; nt0 < 2; ++nt0) {
        const int nt = cq * 2 + nt0;
        v8f c = {};
        #pragma unroll
        for (int kt = 0; kt < 8; ++kt) {
            FragBF a, b;
            const Q16* ap = (const Q16*)(bbuf + (rt * 16 + lm) * 256 + kt * 32 + lh * 8);
            a.q[0] = ap[0]; a.q[1] = ap[2];          // elems 8..15 live at K+16
            b.v = *(const v16bf*)(w2p + (size_t)((u * 8 + kt) * 8 + nt) * 512 + lane * 16);
            c = __builtin_amdgcn_wmma_f32_16x16x32_bf16(false, a.v, false, b.v,
                                                        (short)0, c, false, false);
        }
        const int col = nt * 16 + lm;
        #pragma unroll
        for (int j = 0; j < 8; ++j)
            fbuf[(rt * 16 + lh * 8 + j) * 128 + col] = c[j] + b2[u * 128 + col];
    }
    __syncthreads();
    for (int rr = 0; rr < 4; ++rr) {                 // LN over 128
        const int row = wave * 4 + rr;
        float vv[4];
        #pragma unroll
        for (int e = 0; e < 4; ++e) vv[e] = fbuf[row * 128 + lane * 4 + e];
        float s = 0.f, s2 = 0.f;
        #pragma unroll
        for (int e = 0; e < 4; ++e) { s += vv[e]; s2 += vv[e] * vv[e]; }
        #pragma unroll
        for (int off = 16; off > 0; off >>= 1) {
            s += __shfl_xor(s, off, 32); s2 += __shfl_xor(s2, off, 32);
        }
        const float mean = s * (1.f / 128.f);
        const float rstd = rsqrtf(s2 * (1.f / 128.f) - mean * mean + 1e-5f);
        #pragma unroll
        for (int e = 0; e < 4; ++e) {
            const int col = lane * 4 + e;
            const float t = (vv[e] - mean) * rstd * g2[u * 128 + col] + be2[u * 128 + col];
            bbuf[row * 128 + col] = (__bf16)fmaxf(t, 0.f);
        }
    }
    __syncthreads();

    // ---- stage 3: enc = h2 @ W3 + b3  [32 x 64] -> LDS + global
    {
        const int nt = cq;
        v8f c = {};
        #pragma unroll
        for (int kt = 0; kt < 4; ++kt) {
            FragBF a, b;
            const Q16* ap = (const Q16*)(bbuf + (rt * 16 + lm) * 128 + kt * 32 + lh * 8);
            a.q[0] = ap[0]; a.q[1] = ap[2];
            b.v = *(const v16bf*)(w3p + (size_t)((u * 4 + kt) * 4 + nt) * 512 + lane * 16);
            c = __builtin_amdgcn_wmma_f32_16x16x32_bf16(false, a.v, false, b.v,
                                                        (short)0, c, false, false);
        }
        const int col = nt * 16 + lm;
        #pragma unroll
        for (int j = 0; j < 8; ++j) {
            const int row = rt * 16 + lh * 8 + j;
            const float val = c[j] + b3[u * 64 + col];
            fbuf[row * 64 + col] = val;
            encG[((size_t)(b0 + row) * U_ + u) * E_ + col] = val;
        }
    }
    __syncthreads();

    // sum(enc^2) per row
    for (int rr = 0; rr < 4; ++rr) {
        const int row = wave * 4 + rr;
        const float e0 = fbuf[row * 64 + lane * 2];
        const float e1 = fbuf[row * 64 + lane * 2 + 1];
        float s2 = e0 * e0 + e1 * e1;
        #pragma unroll
        for (int off = 16; off > 0; off >>= 1) s2 += __shfl_xor(s2, off, 32);
        if (lane == 0) ssbuf[row] = s2;
    }
    __syncthreads();

    // t = enc@Wr + br + tanh(dm*ss)*enc
    const float dmu = dm[u];
    for (int i = 0; i < 8; ++i) {
        const int o = i * 256 + tid;
        const int row = o >> 6, col = o & 63;
        float acc = br[u * 64 + col];
        for (int e = 0; e < 64; ++e)
            acc += fbuf[row * 64 + e] * Wr[(size_t)u * 4096 + e * 64 + col];
        const float dmi = tanhf(dmu * ssbuf[row]);
        tbuf[row * 64 + col] = acc + dmi * fbuf[row * 64 + col];
    }
    __syncthreads();
    for (int i = 0; i < 4; ++i) {                    // a1 = relu(t@Wa1+ba1)
        const int o = i * 256 + tid;
        const int row = o >> 5, col = o & 31;
        float acc = ba1[u * 32 + col];
        for (int e = 0; e < 64; ++e)
            acc += tbuf[row * 64 + e] * Wa1[(size_t)u * 2048 + e * 32 + col];
        a1buf[row * 32 + col] = fmaxf(acc, 0.f);
    }
    __syncthreads();
    for (int i = 0; i < 2; ++i) {                    // a2 = relu(a1@Wa2+ba2)
        const int o = i * 256 + tid;
        const int row = o >> 4, col = o & 15;
        float acc = ba2[u * 16 + col];
        for (int e = 0; e < 32; ++e)
            acc += a1buf[row * 32 + e] * Wa2[(size_t)u * 512 + e * 16 + col];
        a2buf[row * 16 + col] = fmaxf(acc, 0.f);
    }
    __syncthreads();
    if (tid < 32) {                                  // scores = sigmoid(a2@Wa3+ba3)
        float acc = ba3[u];
        for (int e = 0; e < 16; ++e) acc += a2buf[tid * 16 + e] * Wa3[u * 16 + e];
        scoresG[(size_t)(b0 + tid) * U_ + u] = 1.f / (1.f + __expf(-acc));
    }
}

// ---------------- attention over universe axis: only column means of softmax needed ----
__global__ __launch_bounds__(256)
void attn_kernel(const float* __restrict__ encG, const float* __restrict__ Wqkv,
                 const float* __restrict__ bqkv, float* __restrict__ colmeanG)
{
    extern __shared__ char smem[];
    float*  kbuf   = (float*)smem;                  // [128][64]: enc first, then k
    __bf16* qbuf   = (__bf16*)(smem + 32768);       // [128][64] bf16
    float*  colAcc = (float*)(smem + 49152);        // [128]
    const int b = blockIdx.x;
    const int tid = threadIdx.x;

    for (int i = 0; i < 32; ++i) {                  // stage enc into LDS
        const int idx = i * 256 + tid;
        kbuf[idx] = encG[(size_t)b * 8192 + idx];
    }
    if (tid < 128) colAcc[tid] = 0.f;
    __syncthreads();
    for (int i = 0; i < 32; ++i) {                  // q = enc @ Wq^T + bq (bf16 store)
        const int o = i * 256 + tid;
        const int uu = o >> 6, f = o & 63;
        float acc = bqkv[f];
        for (int e = 0; e < 64; ++e) acc += kbuf[uu * 64 + e] * Wqkv[f * 64 + e];
        qbuf[o] = (__bf16)acc;
    }
    __syncthreads();
    for (int i = 0; i < 32; ++i) {                  // k (reads global enc, reuses kbuf)
        const int o = i * 256 + tid;
        const int uu = o >> 6, f = o & 63;
        float acc = bqkv[64 + f];
        const float* ep = encG + (size_t)b * 8192 + uu * 64;
        for (int e = 0; e < 64; ++e) acc += ep[e] * Wqkv[(64 + f) * 64 + e];
        kbuf[o] = acc;
    }
    __syncthreads();
    const float scale = 0.35355339059327373f;       // 1/sqrt(8)
    for (int i = 0; i < 4; ++i) {                   // 1024 (head,query) rows
        const int r = i * 256 + tid;
        const int h = r >> 7, qq = r & 127;
        float qv[8];
        #pragma unroll
        for (int d = 0; d < 8; ++d) qv[d] = (float)qbuf[qq * 64 + h * 8 + d];
        float m = -1e30f, ssum = 0.f;
        for (int kk = 0; kk < 128; ++kk) {          // pass 1: online max/sum
            const float* kp = &kbuf[kk * 64 + h * 8];
            float dacc = 0.f;
            #pragma unroll
            for (int d = 0; d < 8; ++d) dacc += qv[d] * kp[d];
            dacc *= scale;
            if (dacc > m) { ssum = ssum * __expf(m - dacc) + 1.f; m = dacc; }
            else          { ssum += __expf(dacc - m); }
        }
        const float inv = 1.f / ssum;
        for (int kk = 0; kk < 128; ++kk) {          // pass 2: column accumulation
            const float* kp = &kbuf[kk * 64 + h * 8];
            float dacc = 0.f;
            #pragma unroll
            for (int d = 0; d < 8; ++d) dacc += qv[d] * kp[d];
            atomicAdd(&colAcc[kk], __expf(dacc * scale - m) * inv);
        }
    }
    __syncthreads();
    if (tid < 128)
        colmeanG[(size_t)b * 128 + tid] = colAcc[tid] * (1.f / 1024.f);  // /(H*U)
}

// ---------------- consensus fusion + statistics + output assembly ----------------------
__global__ __launch_bounds__(128)
void final_kernel(const float* __restrict__ scoresG, const float* __restrict__ colmeanG,
                  const float* __restrict__ Wc1, const float* __restrict__ bc1,
                  const float* __restrict__ Wc2, const float* __restrict__ bc2,
                  const float* __restrict__ Wc3, const float* __restrict__ bc3,
                  float* __restrict__ out)
{
    __shared__ float sL[128], c1L[64], c2L[32], consS;
    __shared__ float rA[128], rB[128], rC[128], rD[128];
    const int b = blockIdx.x, t = threadIdx.x;
    sL[t] = scoresG[(size_t)b * 128 + t];
    __syncthreads();
    if (t < 64) {
        float acc = bc1[t];
        for (int e = 0; e < 128; ++e) acc += sL[e] * Wc1[e * 64 + t];
        c1L[t] = fmaxf(acc, 0.f);
    }
    __syncthreads();
    if (t < 32) {
        float acc = bc2[t];
        for (int e = 0; e < 64; ++e) acc += c1L[e] * Wc2[e * 32 + t];
        c2L[t] = fmaxf(acc, 0.f);
    }
    __syncthreads();
    if (t == 0) {
        float acc = bc3[0];
        for (int e = 0; e < 32; ++e) acc += c2L[e] * Wc3[e];
        consS = 1.f / (1.f + __expf(-acc));
    }
    __syncthreads();
    const float sv = sL[t];
    rA[t] = sv;
    rB[t] = sv * sv;
    rC[t] = (sv > 0.1f) ? 1.f : 0.f;
    rD[t] = sv * colmeanG[(size_t)b * 128 + t];
    __syncthreads();
    for (int s = 64; s > 0; s >>= 1) {
        if (t < s) { rA[t] += rA[t + s]; rB[t] += rB[t + s]; rC[t] += rC[t + s]; rD[t] += rD[t + s]; }
        __syncthreads();
    }
    if (t == 0) {
        const float cons = consS;
        const float mean = rA[0] * (1.f / 128.f);
        float var = (rB[0] - 128.f * mean * mean) * (1.f / 127.f);   // unbiased
        var = fmaxf(var, 0.f);
        out[b]        = rD[0] * cons;                       // mva
        out[1024 + b] = (cons > 0.95f) ? 1.f : 0.f;         // collapse
        out[2048 + b] = 1.f - sqrtf(var) / (mean + 1e-8f);  // agree
        out[3072 + b] = cons;                               // cons
        out[4097 + b] = rC[0];                              // n_active
        if (b == 0) out[4096] = 1.f / 128.f;                // attn_w.mean() == 1/U exactly
    }
}

extern "C" void kernel_launch(void* const* d_in, const int* in_sizes, int n_in,
                              void* d_out, int out_size, void* d_ws, size_t ws_size,
                              hipStream_t stream) {
    const float* x    = (const float*)d_in[0];
    const float* fs   = (const float*)d_in[1];
    const float* dm   = (const float*)d_in[2];
    const float* W1   = (const float*)d_in[3];
    const float* b1   = (const float*)d_in[4];
    const float* g1   = (const float*)d_in[5];
    const float* be1  = (const float*)d_in[6];
    const float* W2   = (const float*)d_in[7];
    const float* b2   = (const float*)d_in[8];
    const float* g2   = (const float*)d_in[9];
    const float* be2  = (const float*)d_in[10];
    const float* W3   = (const float*)d_in[11];
    const float* b3   = (const float*)d_in[12];
    const float* Wr   = (const float*)d_in[13];
    const float* br   = (const float*)d_in[14];
    const float* Wa1  = (const float*)d_in[15];
    const float* ba1  = (const float*)d_in[16];
    const float* Wa2  = (const float*)d_in[17];
    const float* ba2  = (const float*)d_in[18];
    const float* Wa3  = (const float*)d_in[19];
    const float* ba3  = (const float*)d_in[20];
    const float* Wqkv = (const float*)d_in[21];
    const float* bqkv = (const float*)d_in[22];
    // d_in[23]=Wo, d_in[24]=bo: dead (outputs do not depend on them)
    const float* Wc1  = (const float*)d_in[25];
    const float* bc1  = (const float*)d_in[26];
    const float* Wc2  = (const float*)d_in[27];
    const float* bc2  = (const float*)d_in[28];
    const float* Wc3  = (const float*)d_in[29];
    const float* bc3  = (const float*)d_in[30];

    char* ws = (char*)d_ws;
    size_t off = 0;
    __bf16* xpack   = (__bf16*)(ws + off); off += (size_t)64 * 8 * 512 * 2;        // 512 KB
    __bf16* w1p     = (__bf16*)(ws + off); off += (size_t)128 * 8 * 16 * 512 * 2;  // 16 MB
    __bf16* w2p     = (__bf16*)(ws + off); off += (size_t)128 * 8 * 8 * 512 * 2;   // 8 MB
    __bf16* w3p     = (__bf16*)(ws + off); off += (size_t)128 * 4 * 4 * 512 * 2;   // 2 MB
    float*  encG    = (float*)(ws + off);  off += (size_t)1024 * 128 * 64 * 4;     // 32 MB
    float*  scoresG = (float*)(ws + off);  off += (size_t)1024 * 128 * 4;          // 512 KB
    float*  colmeanG= (float*)(ws + off);  off += (size_t)1024 * 128 * 4;          // 512 KB

    pack_x_kernel<<<1024, 256, 0, stream>>>(x, xpack);
    pack_w_kernel<<<dim3(256, 128), 256, 0, stream>>>(W1, w1p, 256, 256);
    pack_w_kernel<<<dim3(128, 128), 256, 0, stream>>>(W2, w2p, 256, 128);
    pack_w_kernel<<<dim3(32, 128), 256, 0, stream>>>(W3, w3p, 128, 64);

    universe_kernel<<<dim3(128, 32), 256, 63616, stream>>>(
        fs, dm, b1, g1, be1, b2, g2, be2, b3, Wr, br,
        Wa1, ba1, Wa2, ba2, Wa3, ba3,
        xpack, w1p, w2p, w3p, encG, scoresG);

    attn_kernel<<<1024, 256, 49664, stream>>>(encG, Wqkv, bqkv, colmeanG);

    final_kernel<<<1024, 128, 0, stream>>>(scoresG, colmeanG,
                                           Wc1, bc1, Wc2, bc2, Wc3, bc3,
                                           (float*)d_out);
    (void)in_sizes; (void)n_in; (void)out_size; (void)ws_size;
}
